// QuantizationModule_2336462209596
// MI455X (gfx1250) — matrix operand order repaired and checked by reading, not compiled, not run on hardware
//
#include <hip/hip_runtime.h>

// Gumbel-softmax VQ forward for MI455X (gfx1250, wave32, WMMA).
// Forward value of (hard - sg(probs) + probs) == one_hot(argmax) to within 1 ulp,
// so we compute: idx = argmax(z_g @ W_logits + gumbel(noise)); out = gather(cb, idx) @ W_out + b_out.
// Both GEMMs use v_wmma_f32_16x16x32_bf16 with a bf16 hi/lo split (3 WMMAs) for ~fp32 accuracy.
// A-operands are converted to bf16 hi/lo ONCE during the cooperative LDS fill so the
// WMMA inner loops are pure ds_load_b128 + global_load_b128 + v_wmma.

typedef __bf16 v16bf __attribute__((ext_vector_type(16)));
typedef __bf16 v8bf  __attribute__((ext_vector_type(8)));
typedef __bf16 v4bf  __attribute__((ext_vector_type(4)));
typedef float  v8f   __attribute__((ext_vector_type(8)));

#define GEPS 1e-10f
#define NEG_INF (-3.402823466e38f)

// ---------------------------------------------------------------------------
// Pack a row-major f32 weight [K, ldn] into WMMA-B-fragment-ordered bf16 hi/lo.
// Fragment flat index: ((t*k_steps + s)*32 + lane)*16 + j
//   lane -> n = t*16 + (lane&15), k = s*32 + (j>>3)*16 + (lane>>4)*8 + (j&7)
__global__ void pack_w_frag(const float* __restrict__ W, __bf16* __restrict__ hi,
                            __bf16* __restrict__ lo, int n_tiles, int k_steps, int ldn) {
    int id = blockIdx.x * 256 + threadIdx.x;
    int total = n_tiles * k_steps * 32 * 16;
    if (id >= total) return;
    int j = id & 15;
    int l = (id >> 4) & 31;
    int rest = id >> 9;
    int s = rest % k_steps;
    int t = rest / k_steps;
    int k = s * 32 + ((j >> 3) << 4) + ((l >> 4) << 3) + (j & 7);
    int n = t * 16 + (l & 15);
    float x = W[(size_t)k * ldn + n];
    __bf16 h = (__bf16)x;
    hi[id] = h;
    lo[id] = (__bf16)(x - (float)h);
}

// Plain elementwise f32 -> bf16 hi/lo (codebooks).
__global__ void pack_plain(const float* __restrict__ src, __bf16* __restrict__ hi,
                           __bf16* __restrict__ lo, int n) {
    int id = blockIdx.x * 256 + threadIdx.x;
    if (id >= n) return;
    float x = src[id];
    __bf16 h = (__bf16)x;
    hi[id] = h;
    lo[id] = (__bf16)(x - (float)h);
}

// ---------------------------------------------------------------------------
// Kernel 1: logits GEMM [65536,384]x[384,320] (bf16 hi/lo split) + gumbel + argmax.
// WG = 128 threads (4 waves). All waves share the same 16 rows; N split 5 tiles/wave.
__global__ __launch_bounds__(128)
void gemm1_argmax(const float* __restrict__ z,       // [65536, 384] (= z reshaped)
                  const float* __restrict__ noise,   // [65536, 320]
                  const float* __restrict__ b_logits,// [320]
                  const __bf16* __restrict__ Bh,     // packed frags [20*12*32*16]
                  const __bf16* __restrict__ Bl,
                  int* __restrict__ idx_out) {       // [65536]
    __shared__ __bf16 Ah_s[16 * 392];   // 16 rows x 384 bf16(hi), +8 pad (stride 196 dwords == 4 mod 64)
    __shared__ __bf16 Al_s[16 * 392];
    __shared__ float redv[4][16];
    __shared__ int   redn[4][16];

    const int tid  = threadIdx.x;
    const int wave = tid >> 5;
    const int lane = tid & 31;
    const int m    = lane & 15;
    const int kh   = lane >> 4;
    const int row0 = blockIdx.x * 16;

    // Cooperative load of the 16x384 f32 A block, split to bf16 hi/lo into LDS
    // (conversion done once per element, not once per wave per k-step).
    {
        const float4* src = (const float4*)(z + (size_t)row0 * 384);
        for (int c = tid; c < 1536; c += 128) {
            int r = c / 96, cc = c % 96;
            float4 v = src[r * 96 + cc];
            float xs[4] = {v.x, v.y, v.z, v.w};
            v4bf hv, lv;
#pragma unroll
            for (int j = 0; j < 4; ++j) {
                __bf16 h = (__bf16)xs[j];
                hv[j] = h;
                lv[j] = (__bf16)(xs[j] - (float)h);
            }
            int dd = r * 392 + cc * 4;
            *(v4bf*)&Ah_s[dd] = hv;
            *(v4bf*)&Al_s[dd] = lv;
        }
    }
    __syncthreads();

    v8f acc[5];
#pragma unroll
    for (int t = 0; t < 5; ++t)
#pragma unroll
        for (int i = 0; i < 8; ++i) acc[t][i] = 0.0f;

    for (int ks = 0; ks < 12; ++ks) {
        // A fragment per ISA 16-bit 16x32 layout: j<8 -> k = kh*8+j ; j>=8 -> k = 16+kh*8+(j-8)
        int ab = m * 392 + ks * 32 + kh * 8;
        v8bf h0 = *(const v8bf*)&Ah_s[ab];
        v8bf h1 = *(const v8bf*)&Ah_s[ab + 16];
        v8bf l0 = *(const v8bf*)&Al_s[ab];
        v8bf l1 = *(const v8bf*)&Al_s[ab + 16];
        v16bf Ah, Al;
#pragma unroll
        for (int j = 0; j < 8; ++j) {
            Ah[j] = h0[j]; Ah[j + 8] = h1[j];
            Al[j] = l0[j]; Al[j + 8] = l1[j];
        }
#pragma unroll
        for (int tt = 0; tt < 5; ++tt) {
            int t = wave * 5 + tt;
            size_t fo = ((size_t)(t * 12 + ks) * 32 + lane) * 16;
            v16bf Bhf = *(const v16bf*)&Bh[fo];
            v16bf Blf = *(const v16bf*)&Bl[fo];
            acc[tt] = __builtin_amdgcn_wmma_f32_16x16x32_bf16(false, Ah, false, Bhf, (short)0, acc[tt], false, false);
            acc[tt] = __builtin_amdgcn_wmma_f32_16x16x32_bf16(false, Ah, false, Blf, (short)0, acc[tt], false, false);
            acc[tt] = __builtin_amdgcn_wmma_f32_16x16x32_bf16(false, Al, false, Bhf, (short)0, acc[tt], false, false);
        }
    }

    // Epilogue: + b_logits + gumbel, per-row argmax.
    // C layout: slot i holds (row m = i + 8*kh, col n = tile*16 + (lane&15)).
    float bestv[8]; int bestn[8];
#pragma unroll
    for (int i = 0; i < 8; ++i) { bestv[i] = NEG_INF; bestn[i] = 0; }

#pragma unroll
    for (int tt = 0; tt < 5; ++tt) {
        int n = (wave * 5 + tt) * 16 + m;
        float bl = b_logits[n];
#pragma unroll
        for (int i = 0; i < 8; ++i) {
            int r = row0 + i + 8 * kh;
            float u = noise[(size_t)r * 320 + n];
            float gum = -__logf(-__logf(u + GEPS) + GEPS);
            float val = acc[tt][i] + bl + gum;
            if (val > bestv[i] || (val == bestv[i] && n < bestn[i])) { bestv[i] = val; bestn[i] = n; }
        }
    }
    // Reduce across the 16 lanes that share each row (xor masks stay within half-wave).
#pragma unroll
    for (int off = 1; off < 16; off <<= 1) {
#pragma unroll
        for (int i = 0; i < 8; ++i) {
            float ov = __shfl_xor(bestv[i], off, 32);
            int   on = __shfl_xor(bestn[i], off, 32);
            if (ov > bestv[i] || (ov == bestv[i] && on < bestn[i])) { bestv[i] = ov; bestn[i] = on; }
        }
    }
    if (m == 0) {  // lanes 0 and 16 hold rows 0..7 / 8..15
#pragma unroll
        for (int i = 0; i < 8; ++i) { redv[wave][i + 8 * kh] = bestv[i]; redn[wave][i + 8 * kh] = bestn[i]; }
    }
    __syncthreads();
    if (tid < 16) {
        float bv = redv[0][tid]; int bn = redn[0][tid];
#pragma unroll
        for (int w = 1; w < 4; ++w) {
            float v = redv[w][tid]; int n2 = redn[w][tid];
            if (v > bv || (v == bv && n2 < bn)) { bv = v; bn = n2; }
        }
        idx_out[row0 + tid] = bn;
    }
}

// ---------------------------------------------------------------------------
// Kernel 2: gather codebook rows by idx, GEMM [32768,768]x[768,768] (hi/lo split), + b_out.
// WG = 128 threads (4 waves) on 16 tokens; N=768 split 12 tiles/wave, K = 24 steps.
__global__ __launch_bounds__(128)
void gemm2_out(const int* __restrict__ idx,           // [65536]
               const __bf16* __restrict__ cbh,        // [2*320*384]
               const __bf16* __restrict__ cbl,
               const __bf16* __restrict__ Woh,        // packed frags [48*24*32*16]
               const __bf16* __restrict__ Wol,
               const float* __restrict__ b_out,       // [768]
               float* __restrict__ out) {             // [32768, 768]
    __shared__ __bf16 Ah_s[16 * 776];   // 16 tokens x 768 bf16, +8 pad (stride 388 dwords == 4 mod 64)
    __shared__ __bf16 Al_s[16 * 776];

    const int tid  = threadIdx.x;
    const int wave = tid >> 5;
    const int lane = tid & 31;
    const int m    = lane & 15;
    const int kh   = lane >> 4;
    const int tok0 = blockIdx.x * 16;

    // Gather: 16 rows x 768 elems in 8-element chunks (1536 chunks, hi & lo).
    for (int c = tid; c < 1536; c += 128) {
        int r = c / 96, cc = c % 96;
        int g = cc / 48;
        int id2 = idx[(tok0 + r) * 2 + g];
        size_t so = ((size_t)(g * 320 + id2)) * 384 + (size_t)(cc - g * 48) * 8;
        int    dd = r * 776 + cc * 8;
        *(v8bf*)&Ah_s[dd] = *(const v8bf*)&cbh[so];
        *(v8bf*)&Al_s[dd] = *(const v8bf*)&cbl[so];
    }
    __syncthreads();

    v8f acc[12];
#pragma unroll
    for (int t = 0; t < 12; ++t)
#pragma unroll
        for (int i = 0; i < 8; ++i) acc[t][i] = 0.0f;

    for (int ks = 0; ks < 24; ++ks) {
        int ab = m * 776 + ks * 32 + kh * 8;
        v8bf h0 = *(const v8bf*)&Ah_s[ab];
        v8bf h1 = *(const v8bf*)&Ah_s[ab + 16];
        v8bf l0 = *(const v8bf*)&Al_s[ab];
        v8bf l1 = *(const v8bf*)&Al_s[ab + 16];
        v16bf Ah, Al;
#pragma unroll
        for (int j = 0; j < 8; ++j) {
            Ah[j] = h0[j]; Ah[j + 8] = h1[j];
            Al[j] = l0[j]; Al[j + 8] = l1[j];
        }
#pragma unroll
        for (int tt = 0; tt < 12; ++tt) {
            int t = wave * 12 + tt;
            size_t fo = ((size_t)(t * 24 + ks) * 32 + lane) * 16;
            v16bf Bhf = *(const v16bf*)&Woh[fo];
            v16bf Blf = *(const v16bf*)&Wol[fo];
            acc[tt] = __builtin_amdgcn_wmma_f32_16x16x32_bf16(false, Ah, false, Bhf, (short)0, acc[tt], false, false);
            acc[tt] = __builtin_amdgcn_wmma_f32_16x16x32_bf16(false, Ah, false, Blf, (short)0, acc[tt], false, false);
            acc[tt] = __builtin_amdgcn_wmma_f32_16x16x32_bf16(false, Al, false, Bhf, (short)0, acc[tt], false, false);
        }
    }

    // Epilogue: +bias, coalesced f32 stores (lanes 0..15 cover 16 consecutive n).
#pragma unroll
    for (int tt = 0; tt < 12; ++tt) {
        int n = (wave * 12 + tt) * 16 + m;
        float bo = b_out[n];
#pragma unroll
        for (int i = 0; i < 8; ++i) {
            out[(size_t)(tok0 + i + 8 * kh) * 768 + n] = acc[tt][i] + bo;
        }
    }
}

// ---------------------------------------------------------------------------
extern "C" void kernel_launch(void* const* d_in, const int* in_sizes, int n_in,
                              void* d_out, int out_size, void* d_ws, size_t ws_size,
                              hipStream_t stream) {
    (void)in_sizes; (void)n_in; (void)out_size; (void)ws_size;
    const float* z         = (const float*)d_in[0]; // [16,2048,768] -> [65536,384]
    const float* noise     = (const float*)d_in[1]; // [65536,320]
    const float* W_logits  = (const float*)d_in[2]; // [384,320]
    const float* b_logits  = (const float*)d_in[3]; // [320]
    const float* codebooks = (const float*)d_in[4]; // [2,320,384]
    const float* W_out     = (const float*)d_in[5]; // [768,768]
    const float* b_out     = (const float*)d_in[6]; // [768]
    float* out = (float*)d_out;

    char* ws = (char*)d_ws;
    __bf16* Wl_hi = (__bf16*)(ws + 0);        // 122880 elems
    __bf16* Wl_lo = (__bf16*)(ws + 245760);
    __bf16* Wo_hi = (__bf16*)(ws + 491520);   // 589824 elems
    __bf16* Wo_lo = (__bf16*)(ws + 1671168);
    __bf16* cb_hi = (__bf16*)(ws + 2850816);  // 245760 elems
    __bf16* cb_lo = (__bf16*)(ws + 3342336);
    int*    idx   = (int*)   (ws + 3833856);  // 65536 ints   (total ws ~4.1 MB)

    pack_w_frag<<<480,  256, 0, stream>>>(W_logits, Wl_hi, Wl_lo, 20, 12, 320);
    pack_w_frag<<<2304, 256, 0, stream>>>(W_out,    Wo_hi, Wo_lo, 48, 24, 768);
    pack_plain <<<960,  256, 0, stream>>>(codebooks, cb_hi, cb_lo, 245760);
    gemm1_argmax<<<4096, 128, 0, stream>>>(z, noise, b_logits, Wl_hi, Wl_lo, idx);
    gemm2_out  <<<2048, 128, 0, stream>>>(idx, cb_hi, cb_lo, Wo_hi, Wo_lo, b_out, out);
}
